// ScaledDotProductAttention_403726926205
// MI455X (gfx1250) — matrix operand order repaired and checked by reading, not compiled, not run on hardware
//
#include <hip/hip_runtime.h>
#include <hip/hip_bf16.h>

typedef __attribute__((ext_vector_type(16))) __bf16 v16bf;
typedef __attribute__((ext_vector_type(8)))  float  v8f;
typedef __attribute__((ext_vector_type(16))) unsigned short v16us;

struct alignas(16) U4 { unsigned int x, y, z, w; };
union FragU { U4 q[2]; v16us us; v16bf bf; };

static __device__ __forceinline__ unsigned short f2bf(float f) {
  unsigned int u = __builtin_bit_cast(unsigned int, f);
  return (unsigned short)((u + 0x7FFFu + ((u >> 16) & 1u)) >> 16);  // RNE
}

#define WMMA_BF16(A, B, C) \
  __builtin_amdgcn_wmma_f32_16x16x32_bf16(false, (A), false, (B), (short)0, (C), false, false)

#if __has_builtin(__builtin_amdgcn_exp2f)
#define EXPX(x) __builtin_amdgcn_exp2f(x)     // raw v_exp_f32 (base 2)
#define K_LOG2E 1.4426950408889634f
#else
#define EXPX(x) __expf(x)
#define K_LOG2E 1.0f
#endif

constexpr int   kS      = 2048;
constexpr int   kD      = 64;
constexpr int   kBH     = 32;
constexpr float kQScale = 0.125f * K_LOG2E;            // 1/sqrt(64) * base-change
constexpr float kMaskS  = -4294967296.0f * K_LOG2E;    // MASK_NUM in exp2 domain
constexpr int   KP      = 80;   // sK row pitch (ushorts), 16B-aligned rows
constexpr int   VP      = 40;   // sV row pitch (ushorts), 16B-aligned rows

__global__ __launch_bounds__(256) void fa_fwd(
    const float* __restrict__ Q, const float* __restrict__ Km,
    const float* __restrict__ Vm, const int* __restrict__ Mk,
    float* __restrict__ Out)
{
  __shared__ alignas(16) unsigned short sK[2][32 * KP];  // K tile bf16 [key][d]
  __shared__ alignas(16) unsigned short sV[2][64 * VP];  // V tile bf16 [d][key] (transposed)
  __shared__ float sMask[2][32];

  const int bh    = blockIdx.y;
  const int qBase = blockIdx.x * 128;
  const int tid   = threadIdx.x;
  const int wv    = tid >> 5;          // wave id: q sub-tile
  const int lane  = tid & 31;
  const int n     = lane & 15;         // N column (q for S^T, d for O)
  const int c     = lane >> 4;         // half-wave select
  const int qRow  = qBase + 16 * wv + n;
  const int* mrow = Mk + (size_t)(bh >> 3) * kS;   // masks tiled by heads (BH/B = 8)
  const size_t baseKV = (size_t)bh * kS * kD;

  // staging thread mapping: 256 threads cover 32 keys x (8 d-groups of 8)
  const int sKey = tid >> 3;
  const int sD0  = (tid & 7) << 3;

  // ---- Q as B-fragments of S^T = K * Q^T, with softmax scale pre-folded ----
  const float* qp = Q + ((size_t)bh * kS + qRow) * kD;
  FragU bq0, bq1;
#pragma unroll
  for (int e = 0; e < 16; ++e) {
    bq0.us[e] = f2bf(qp[8 * c + e + (e & 8)] * kQScale);        // e<8: d=e+8c ; e>=8: d=e+8+8c
    bq1.us[e] = f2bf(qp[32 + 8 * c + e + (e & 8)] * kQScale);   // same, offset by 32
  }

  const v8f vzero = {0.f, 0.f, 0.f, 0.f, 0.f, 0.f, 0.f, 0.f};
  v8f acc[4] = {vzero, vzero, vzero, vzero};  // O[q = j+8c][d = 16f + n]
  float mrun = -3.0e38f;
  float lrun = 0.0f;

  const int nkb = (qBase + 128) >> 5;  // causal: uniform trip count per block

  // ---- prologue: global loads for tile 0 land in registers ----
  float krg[8], vrg[8], mvf;
  {
    const float* kp = Km + baseKV + (size_t)sKey * kD + sD0;
    const float* vp = Vm + baseKV + (size_t)sKey * kD + sD0;
#pragma unroll
    for (int i = 0; i < 8; ++i) { krg[i] = kp[i]; vrg[i] = vp[i]; }
    mvf = (float)mrow[tid & 31] * kMaskS;
  }

  for (int kb = 0; kb < nkb; ++kb) {
    const int k0  = kb << 5;
    const int buf = kb & 1;

    // ---- drain in-flight registers into LDS buffer `buf` (f32 -> bf16) ----
    {
      U4 pk;
      pk.x = (unsigned)f2bf(krg[0]) | ((unsigned)f2bf(krg[1]) << 16);
      pk.y = (unsigned)f2bf(krg[2]) | ((unsigned)f2bf(krg[3]) << 16);
      pk.z = (unsigned)f2bf(krg[4]) | ((unsigned)f2bf(krg[5]) << 16);
      pk.w = (unsigned)f2bf(krg[6]) | ((unsigned)f2bf(krg[7]) << 16);
      *(U4*)&sK[buf][sKey * KP + sD0] = pk;           // ds_store_b128
#pragma unroll
      for (int i = 0; i < 8; ++i)                     // transpose V into [d][key]
        sV[buf][(sD0 + i) * VP + sKey] = f2bf(vrg[i]);
      if (tid < 32) sMask[buf][tid] = mvf;
    }

    // ---- prefetch tile kb+1 into registers (overlaps barrier + compute) ----
    if (kb + 1 < nkb) {
      const int k0n = (kb + 1) << 5;
      const float* kp = Km + baseKV + (size_t)(k0n + sKey) * kD + sD0;
      const float* vp = Vm + baseKV + (size_t)(k0n + sKey) * kD + sD0;
#pragma unroll
      for (int i = 0; i < 8; ++i) { krg[i] = kp[i]; vrg[i] = vp[i]; }
      mvf = (float)mrow[k0n + (tid & 31)] * kMaskS;
    }

    __syncthreads();   // buffer `buf` staged; buffer `buf^1` free to refill next iter

    // ---- S^T = (K tile) x (Q^T): two 16-key subtiles, D=64 split in two ----
    float p[2][8];
    float mt = -3.0e38f;
#pragma unroll
    for (int sub = 0; sub < 2; ++sub) {
      const unsigned short* kr = &sK[buf][(sub * 16 + n) * KP];
      FragU a0, a1;                               // A-frag e<8: d=e+8c ; e>=8: d=e+8+8c
      a0.q[0] = *(const U4*)&kr[8 * c];
      a0.q[1] = *(const U4*)&kr[16 + 8 * c];
      a1.q[0] = *(const U4*)&kr[32 + 8 * c];
      a1.q[1] = *(const U4*)&kr[48 + 8 * c];
      v8f sc = vzero;
      sc = WMMA_BF16(a0.bf, bq0.bf, sc);          // d 0..31
      sc = WMMA_BF16(a1.bf, bq1.bf, sc);          // d 32..63
#pragma unroll
      for (int j = 0; j < 8; ++j) {               // S^T[key=j+8c][q=n]
        const int key = k0 + sub * 16 + j + 8 * c;
        float s = sc[j] + sMask[buf][sub * 16 + j + 8 * c];
        if (key > qRow) s = kMaskS;               // causal
        p[sub][j] = s;
        mt = fmaxf(mt, s);
      }
    }

    // ---- online softmax (exp2 domain); row q=n lives in lanes n and n+16 ----
    mt = fmaxf(mt, __shfl_xor(mt, 16, 32));
    const float mNew = fmaxf(mrun, mt);
    const float corr = EXPX(mrun - mNew);
    mrun = mNew;
    float ps = 0.0f;
#pragma unroll
    for (int sub = 0; sub < 2; ++sub)
#pragma unroll
      for (int j = 0; j < 8; ++j) {
        const float e = EXPX(p[sub][j] - mNew);
        p[sub][j] = e;
        ps += e;
      }
    ps += __shfl_xor(ps, 16, 32);
    lrun = lrun * corr + ps;

    // ---- pack P as A-frag: register-local thanks to transposed score layout ----
    FragU pf;
#pragma unroll
    for (int e = 0; e < 8; ++e) {
      pf.us[e]     = f2bf(p[0][e]);   // keys  8c..8c+7
      pf.us[e + 8] = f2bf(p[1][e]);   // keys 16+8c..23+8c
    }

    // ---- rescale O rows (corr for q=j+8c lives at lane j+8c) ----
#pragma unroll
    for (int j = 0; j < 8; ++j) {
      const float cj = __shfl(corr, j + 8 * c, 32);
      acc[0][j] *= cj; acc[1][j] *= cj; acc[2][j] *= cj; acc[3][j] *= cj;
    }

    // ---- O += P(16x32) x V(32x16) for four d-column groups ----
#pragma unroll
    for (int f = 0; f < 4; ++f) {
      const unsigned short* vr = &sV[buf][(16 * f + n) * VP + 8 * c];
      FragU bv;                                    // B-frag e<8: key = e + 8c ; e>=8: key = e + 8 + 8c
      bv.q[0] = *(const U4*)&vr[0];
      bv.q[1] = *(const U4*)&vr[16];
      acc[f] = WMMA_BF16(pf.bf, bv.bf, acc[f]);
    }
  }

  // ---- normalize and store (nontemporal: written once, never re-read) ----
  const float inv = 1.0f / lrun;
#pragma unroll
  for (int j = 0; j < 8; ++j) {
    const float ij = __shfl(inv, j + 8 * c, 32);
    const int qo = qBase + 16 * wv + j + 8 * c;
    float* op = Out + ((size_t)bh * kS + qo) * kD + n;
    __builtin_nontemporal_store(acc[0][j] * ij, op + 0);
    __builtin_nontemporal_store(acc[1][j] * ij, op + 16);
    __builtin_nontemporal_store(acc[2][j] * ij, op + 32);
    __builtin_nontemporal_store(acc[3][j] * ij, op + 48);
  }
}

extern "C" void kernel_launch(void* const* d_in, const int* in_sizes, int n_in,
                              void* d_out, int out_size, void* d_ws, size_t ws_size,
                              hipStream_t stream) {
  (void)in_sizes; (void)n_in; (void)out_size; (void)d_ws; (void)ws_size;
  const float* Q = (const float*)d_in[0];
  const float* K = (const float*)d_in[1];
  const float* V = (const float*)d_in[2];
  const int*   M = (const int*)d_in[3];
  float* Out = (float*)d_out;
  dim3 grid(kS / 128, kBH);   // 16 q-blocks x 32 batch*heads
  fa_fwd<<<grid, 256, 0, stream>>>(Q, K, V, M, Out);
}
